// GAE_FiLM_36644660970267
// MI455X (gfx1250) — matrix operand order repaired and checked by reading, not compiled, not run on hardware
//
#include <hip/hip_runtime.h>

typedef __attribute__((ext_vector_type(16))) _Float16 v16h;
typedef __attribute__((ext_vector_type(8)))  float    v8f;

#define ZDIM 32
#define HDIM 64
#define BGRAPH 1024

// ---- order-preserving float <-> uint encoding (for atomicMax on floats) ----
__device__ __forceinline__ unsigned enc_f32(float x) {
  unsigned b = __float_as_uint(x);
  return (b & 0x80000000u) ? ~b : (b | 0x80000000u);
}
__device__ __forceinline__ float dec_f32(unsigned u) {
  return (u & 0x80000000u) ? __uint_as_float(u ^ 0x80000000u)
                           : __uint_as_float(~u);
}
// enc(-inf) = ~0xFF800000 = 0x007FFFFF
#define ENC_NEG_INF 0x007FFFFFu

// ---------------------------------------------------------------- init ----
__global__ void k_init(unsigned* segmax, float* denom, float* g_num) {
  int i = blockIdx.x * 256 + threadIdx.x;
  if (i < BGRAPH) { segmax[i] = ENC_NEG_INF; denom[i] = 0.f; }
  if (i < BGRAPH * ZDIM) g_num[i] = 0.f;
}

// ------------------------------------------------- pass 1: gate + segmax ----
__global__ void __launch_bounds__(256) k_gate(const float4* __restrict__ z4,
                                              const float* __restrict__ Wg,
                                              const float* __restrict__ bg,
                                              const int* __restrict__ batch,
                                              float* __restrict__ gate,
                                              unsigned* __restrict__ segmax,
                                              int N) {
  __shared__ float wg[ZDIM];
  __shared__ float bgs;
  if (threadIdx.x < ZDIM) wg[threadIdx.x] = Wg[threadIdx.x];
  if (threadIdx.x == 0) bgs = bg[0];
  __syncthreads();
  int row = blockIdx.x * 256 + threadIdx.x;
  if (row >= N) return;
  const float4* zr = z4 + (size_t)row * (ZDIM / 4);
  float acc = bgs;
#pragma unroll
  for (int q = 0; q < ZDIM / 4; ++q) {
    float4 v = zr[q];
    acc += v.x * wg[4 * q + 0] + v.y * wg[4 * q + 1] +
           v.z * wg[4 * q + 2] + v.w * wg[4 * q + 3];
  }
  gate[row] = acc;
  atomicMax(&segmax[batch[row]], enc_f32(acc));
}

// --------------------------- pass 2: e = exp(gate-max), denom, g_num sums ----
__global__ void __launch_bounds__(256) k_accum(const float4* __restrict__ z4,
                                               const float* __restrict__ gate,
                                               const unsigned* __restrict__ segmax,
                                               const int* __restrict__ batch,
                                               float* __restrict__ denom,
                                               float* __restrict__ g_num,
                                               int N) {
  __shared__ float tile[8][ZDIM][ZDIM + 1];
  int wid = threadIdx.x >> 5, lane = threadIdx.x & 31;
  int row = blockIdx.x * 256 + threadIdx.x;
  int crow = row < N ? row : (N - 1);
  int b = batch[crow];
  float e = 0.f;
  if (row < N) e = __expf(gate[row] - dec_f32(segmax[b]));
  const float4* zr = z4 + (size_t)crow * (ZDIM / 4);
#pragma unroll
  for (int q = 0; q < ZDIM / 4; ++q) {
    float4 v = zr[q];
    tile[wid][lane][4 * q + 0] = e * v.x;
    tile[wid][lane][4 * q + 1] = e * v.y;
    tile[wid][lane][4 * q + 2] = e * v.z;
    tile[wid][lane][4 * q + 3] = e * v.w;
  }
  __syncthreads();
  int b0 = __shfl(b, 0, 32);
  bool uni = __all(b == b0);
  if (uni) {  // fast path: batch is sorted, so ~98% of waves land here
    float es = e;
#pragma unroll
    for (int off = 16; off; off >>= 1) es += __shfl_down(es, off, 32);
    if (lane == 0 && es != 0.f) atomicAdd(&denom[b0], es);
    float cs = 0.f;
#pragma unroll
    for (int r = 0; r < 32; ++r) cs += tile[wid][r][lane];
    atomicAdd(&g_num[(size_t)b0 * ZDIM + lane], cs);
  } else if (row < N) {  // segment boundary inside the wave
    atomicAdd(&denom[b], e);
#pragma unroll
    for (int k = 0; k < ZDIM; ++k)
      atomicAdd(&g_num[(size_t)b * ZDIM + k], tile[wid][lane][k]);
  }
}

// --------------------------------------------- g = g_num/denom (+f16 copy) --
__global__ void k_fin_g(const float* __restrict__ g_num,
                        const float* __restrict__ denom,
                        float* __restrict__ g_out, _Float16* __restrict__ g_h) {
  int i = blockIdx.x * 256 + threadIdx.x;
  if (i >= BGRAPH * ZDIM) return;
  float d = denom[i / ZDIM];
  float v = (d > 0.f) ? g_num[i] / d : 0.f;
  g_out[i] = v;
  g_h[i] = (_Float16)v;
}

// ----------------------------------------------------- weights f32 -> f16 ---
__global__ void k_cvt_w(const float* __restrict__ W1, const float* __restrict__ W2,
                        _Float16* __restrict__ w1h, _Float16* __restrict__ w2h) {
  int i = blockIdx.x * 256 + threadIdx.x;
  if (i < ZDIM * HDIM) w1h[i] = (_Float16)W1[i];
  if (i < HDIM * 2 * ZDIM) w2h[i] = (_Float16)W2[i];
}

// -------------------------- layer 1: h = relu(g@W1 + b1), WMMA 16x16x32 -----
__global__ void __launch_bounds__(256) k_mlp1(const _Float16* __restrict__ g_h,
                                              const _Float16* __restrict__ w1h,
                                              const float* __restrict__ b1,
                                              _Float16* __restrict__ hbuf) {
  int wid = threadIdx.x >> 5, lane = threadIdx.x & 31;
  int t = blockIdx.x * 8 + wid;     // 256 tiles: 64 row-tiles x 4 col-tiles
  int mt = t >> 2, nt = t & 3;
  int half = lane >> 4;             // lane group
  int ml = lane & 15;
  // A (16x32 f16): lanes 0-15 hold K=0..15; lanes 16-31 hold K=16..31
  int kb0 = half ? 16 : 0, kb1 = half ? 24 : 8;
  const _Float16* arow = g_h + (size_t)(mt * 16 + ml) * ZDIM;
  v16h a, bm;
#pragma unroll
  for (int i = 0; i < 8; ++i) { a[i] = arow[kb0 + i]; a[8 + i] = arow[kb1 + i]; }
  // B (32x16 f16): lanes 0-15 K=0..15, lanes 16-31 K=16..31; N = lane%16
  int col = nt * 16 + ml;
  int kb = half ? 16 : 0;
#pragma unroll
  for (int i = 0; i < 16; ++i) bm[i] = w1h[(size_t)(kb + i) * HDIM + col];
  float bias = b1[col];
  v8f c;
#pragma unroll
  for (int r = 0; r < 8; ++r) c[r] = bias;
  c = __builtin_amdgcn_wmma_f32_16x16x32_f16(false, a, false, bm, (short)0, c,
                                             false, false);
#pragma unroll
  for (int r = 0; r < 8; ++r) {
    float v = c[r] > 0.f ? c[r] : 0.f;     // ReLU
    int M = r + half * 8;
    hbuf[(size_t)(mt * 16 + M) * HDIM + col] = (_Float16)v;
  }
}

// -------------------------- layer 2: gb = h@W2 + b2, two WMMAs (K=64) -------
__global__ void __launch_bounds__(256) k_mlp2(const _Float16* __restrict__ hbuf,
                                              const _Float16* __restrict__ w2h,
                                              const float* __restrict__ b2,
                                              float* __restrict__ gb) {
  int wid = threadIdx.x >> 5, lane = threadIdx.x & 31;
  int t = blockIdx.x * 8 + wid;
  int mt = t >> 2, nt = t & 3;
  int half = lane >> 4;
  int ml = lane & 15;
  int kb0 = half ? 16 : 0, kb1 = half ? 24 : 8;
  const _Float16* arow = hbuf + (size_t)(mt * 16 + ml) * HDIM;
  v16h a0, a1, bv0, bv1;
#pragma unroll
  for (int i = 0; i < 8; ++i) {
    a0[i] = arow[kb0 + i];       a0[8 + i] = arow[kb1 + i];
    a1[i] = arow[32 + kb0 + i];  a1[8 + i] = arow[32 + kb1 + i];
  }
  int col = nt * 16 + ml;
  int kb = half ? 16 : 0;
#pragma unroll
  for (int i = 0; i < 16; ++i) {
    bv0[i] = w2h[(size_t)(kb + i) * (2 * ZDIM) + col];
    bv1[i] = w2h[(size_t)(32 + kb + i) * (2 * ZDIM) + col];
  }
  float bias = b2[col];
  v8f c;
#pragma unroll
  for (int r = 0; r < 8; ++r) c[r] = bias;
  c = __builtin_amdgcn_wmma_f32_16x16x32_f16(false, a0, false, bv0, (short)0, c,
                                             false, false);
  c = __builtin_amdgcn_wmma_f32_16x16x32_f16(false, a1, false, bv1, (short)0, c,
                                             false, false);
#pragma unroll
  for (int r = 0; r < 8; ++r) {
    int M = r + half * 8;
    gb[(size_t)(mt * 16 + M) * (2 * ZDIM) + col] = c[r];
  }
}

// ----------------------- pass 3: FiLM z_mod = z*(1+gamma[b]) + beta[b] ------
__global__ void __launch_bounds__(256) k_film(const float4* __restrict__ z4,
                                              const int* __restrict__ batch,
                                              const float* __restrict__ gb,
                                              float4* __restrict__ out4, int N) {
  int row = blockIdx.x * 256 + threadIdx.x;
  if (row >= N) return;
  int b = batch[row];
  const float4* gr = (const float4*)(gb + (size_t)b * (2 * ZDIM));
  const float4* zr = z4 + (size_t)row * (ZDIM / 4);
  float4* orow = out4 + (size_t)row * (ZDIM / 4);
#pragma unroll
  for (int q = 0; q < ZDIM / 4; ++q) {
    float4 v = zr[q];
    float4 ga = gr[q];                 // gamma
    float4 be = gr[ZDIM / 4 + q];      // beta
    float4 o;
    o.x = v.x * (1.f + ga.x) + be.x;
    o.y = v.y * (1.f + ga.y) + be.y;
    o.z = v.z * (1.f + ga.z) + be.z;
    o.w = v.w * (1.f + ga.w) + be.w;
    orow[q] = o;
  }
}

// ---------------------------------------------------------------------------
extern "C" void kernel_launch(void* const* d_in, const int* in_sizes, int n_in,
                              void* d_out, int out_size, void* d_ws, size_t ws_size,
                              hipStream_t stream) {
  const float* z  = (const float*)d_in[0];
  const float* Wg = (const float*)d_in[1];
  const float* bg = (const float*)d_in[2];
  const float* W1 = (const float*)d_in[3];
  const float* b1 = (const float*)d_in[4];
  const float* W2 = (const float*)d_in[5];
  const float* b2 = (const float*)d_in[6];
  const int* batch = (const int*)d_in[7];
  const int N = in_sizes[0] / ZDIM;   // 2,000,000

  // workspace carve-out (4B items first to keep 16B alignment)
  char* ws = (char*)d_ws;
  float*    gate   = (float*)ws;    ws += (size_t)N * 4;
  unsigned* segmax = (unsigned*)ws; ws += (size_t)BGRAPH * 4;
  float*    denom  = (float*)ws;    ws += (size_t)BGRAPH * 4;
  float*    g_num  = (float*)ws;    ws += (size_t)BGRAPH * ZDIM * 4;
  float*    gbbuf  = (float*)ws;    ws += (size_t)BGRAPH * 2 * ZDIM * 4;
  _Float16* g_h    = (_Float16*)ws; ws += (size_t)BGRAPH * ZDIM * 2;
  _Float16* hbuf   = (_Float16*)ws; ws += (size_t)BGRAPH * HDIM * 2;
  _Float16* w1h    = (_Float16*)ws; ws += (size_t)ZDIM * HDIM * 2;
  _Float16* w2h    = (_Float16*)ws; ws += (size_t)HDIM * 2 * ZDIM * 2;

  float* zmod = (float*)d_out;
  float* gout = (float*)d_out + (size_t)N * ZDIM;

  const int nblk = (N + 255) / 256;

  k_init<<<(BGRAPH * ZDIM + 255) / 256, 256, 0, stream>>>(segmax, denom, g_num);
  k_gate<<<nblk, 256, 0, stream>>>((const float4*)z, Wg, bg, batch, gate, segmax, N);
  k_accum<<<nblk, 256, 0, stream>>>((const float4*)z, gate, segmax, batch, denom,
                                    g_num, N);
  k_fin_g<<<(BGRAPH * ZDIM + 255) / 256, 256, 0, stream>>>(g_num, denom, gout, g_h);
  k_cvt_w<<<(HDIM * 2 * ZDIM + 255) / 256, 256, 0, stream>>>(W1, W2, w1h, w2h);
  k_mlp1<<<32, 256, 0, stream>>>(g_h, w1h, b1, hbuf);   // 256 wave-tiles
  k_mlp2<<<32, 256, 0, stream>>>(hbuf, w2h, b2, gbbuf); // 256 wave-tiles
  k_film<<<nblk, 256, 0, stream>>>((const float4*)z, batch, gbbuf,
                                   (float4*)zmod, N);
}
